// RALALinearAttention_86363202388358
// MI455X (gfx1250) — compile-verified, matrix operand order checked by reading, not run on hardware
//
#include <hip/hip_runtime.h>
#include <hip/hip_bf16.h>

// ---------------------------------------------------------------------------
// Problem constants (from reference): B=16, Himg=Wimg=64, C=256, NH=8, HD=32
// ---------------------------------------------------------------------------
#define BB    16
#define HIMG  64
#define WIMG  64
#define CC    256
#define NHEAD 8
#define HD    32
#define NN    4096              // Himg*Wimg
#define MROWS (BB * NN)         // 65536
#define QKVO_LD (4 * CC)        // 1024

typedef __bf16 v16bf __attribute__((ext_vector_type(16)));
typedef float  v8f   __attribute__((ext_vector_type(8)));
typedef unsigned int u32x4 __attribute__((ext_vector_type(4)));
typedef int          i32x8 __attribute__((ext_vector_type(8)));
typedef int          i32x4 __attribute__((ext_vector_type(4)));

union FragU { v16bf v; unsigned int u[8]; };

__device__ __forceinline__ float elu1(float x) {
    return x > 0.f ? x + 1.f : __expf(x);
}

// ---------------------------------------------------------------------------
// TDM: async 2D bf16 tile load global -> LDS (Tensor Data Mover, gfx1250).
// Descriptor bit layout per CDNA5 ISA 8.3/8.4 (D# group0 128b, group1 256b).
// tile_w elements per row land contiguously in LDS; pad fields insert
// (pad_amount_code+1) DWORDs after each 2^(pad_interval_code+1) DWORDs.
// This toolchain exposes the 6-arg builtin:
//   (u32x4 g0, i32x8 g1, i32x4 g2, i32x4 g3, i32x8 extra, i32 cpol)
// Groups 2/3 (and the extra group) are zero for 2D tensors.
// ---------------------------------------------------------------------------
__device__ __forceinline__ void tdm_load_tile_bf16(
    unsigned lds_byte_off, unsigned long long gaddr,
    unsigned tensor_w, unsigned tensor_h,
    unsigned tile_w, unsigned tile_h,
    unsigned long long row_stride_elems,
    unsigned pad_interval_code, unsigned pad_amount_code)
{
    u32x4 g0;
    g0[0] = 1u;                                            // count=1 (valid D#)
    g0[1] = lds_byte_off;                                  // lds_addr
    g0[2] = (unsigned)(gaddr & 0xffffffffu);               // global_addr[31:0]
    g0[3] = (unsigned)((gaddr >> 32) & 0x01ffffffu)        // global_addr[56:32]
          | (2u << 30);                                    // type=2 ("image")
    i32x8 g1;
    unsigned d0 = (1u << 16);                              // data_size=1 -> 2 bytes
    if (pad_amount_code | pad_interval_code)
        d0 |= (1u << 20)                                   // pad_enable
            | (pad_interval_code << 22)
            | (pad_amount_code << 25);
    g1[0] = (int)d0;                                       // wg_mask=0 (not in cluster)
    g1[1] = (int)((tensor_w & 0xffffu) << 16);             // tensor_dim0[15:0]
    g1[2] = (int)(((tensor_w >> 16) & 0xffffu)
          | ((tensor_h & 0xffffu) << 16));                 // dim0[31:16] | dim1[15:0]
    g1[3] = (int)(((tensor_h >> 16) & 0xffffu)
          | ((tile_w & 0xffffu) << 16));                   // dim1[31:16] | tile_dim0
    g1[4] = (int)(tile_h & 0xffffu);                       // tile_dim1 | tile_dim2=0
    g1[5] = (int)(unsigned)(row_stride_elems & 0xffffffffu); // dim0_stride[31:0]
    g1[6] = (int)((row_stride_elems >> 32) & 0xffffu);     // dim0_stride[47:32]
    g1[7] = 0;                                             // dim1_stride (unused, 2D)
    i32x4 z4 = {0, 0, 0, 0};
    i32x8 z8 = {0, 0, 0, 0, 0, 0, 0, 0};
    __builtin_amdgcn_tensor_load_to_lds(g0, g1, z4, z4, z8, 0);  // TENSORcnt++
}

// ---------------------------------------------------------------------------
// f32 -> bf16 conversion
// ---------------------------------------------------------------------------
__global__ void cvt_bf16_kernel(const float* __restrict__ in,
                                __bf16* __restrict__ out, int n) {
    int i = blockIdx.x * blockDim.x + threadIdx.x;
    if (i < n) out[i] = (__bf16)in[i];
}

// ---------------------------------------------------------------------------
// bf16 WMMA GEMM: C[M,N] = A[M,K] * B[K,N] + bias[N]   (fp32 accumulate)
// Block tile 128(M) x 64(N), K-step 32. 256 threads = 8 waves.
// A tile fed by TDM (tensor_load_to_lds) with LDS padding to 40-elem rows,
// double buffered so the DMA for tile k+1 overlaps the WMMAs of tile k.
// Wave (wm in 0..3, wn in 0..1) computes a 32x32 sub-tile = 4 WMMA tiles.
// ---------------------------------------------------------------------------
__global__ __launch_bounds__(256) void gemm_bf16_wmma(
    const __bf16* __restrict__ A, const __bf16* __restrict__ Bw,
    const float* __restrict__ bias, float* __restrict__ C,
    int M, int N, int K)
{
    __shared__ __align__(16) __bf16 As[2][128 * 40];  // [m][k] rows, stride 40 (TDM pad)
    __shared__ __align__(16) __bf16 Bs[2][64 * 40];   // [n][k] (transposed), stride 40

    const int tid  = threadIdx.x;
    const int lane = tid & 31;
    const int wave = tid >> 5;
    const int wm   = wave & 3;        // 0..3 -> M offset wm*32
    const int wn   = wave >> 2;       // 0..1 -> N offset wn*32
    const int half = lane >> 4;       // 0/1
    const int l16  = lane & 15;

    const int m0 = blockIdx.y * 128;
    const int n0 = blockIdx.x * 64;
    const unsigned long long abase = (unsigned long long)(uintptr_t)A;

    v8f acc[2][2];
#pragma unroll
    for (int i = 0; i < 2; i++)
#pragma unroll
        for (int j = 0; j < 2; j++)
#pragma unroll
            for (int r = 0; r < 8; r++) acc[i][j][r] = 0.f;

    // ---- helpers -----------------------------------------------------------
    auto issue_A = [&](int buf, int k0) {
        // 128 x 32 bf16 tile at (m0, k0); rows padded 16+4 DWORDs -> stride 40
        unsigned lds_off = (unsigned)(uintptr_t)&As[buf][0];
        unsigned long long gaddr = abase + ((unsigned long long)m0 * K + k0) * 2ull;
        tdm_load_tile_bf16(lds_off, gaddr,
                           (unsigned)(K - k0), 128u,    // tensor dims (OOB bounds)
                           32u, 128u,                   // tile dims
                           (unsigned long long)K,       // row stride (elems)
                           3u, 3u);                     // 16 DW rows + 4 DW pad
    };
    auto fill_B = [&](int buf, int k0) {
        int r = tid >> 3;            // k row 0..31
        int c = (tid & 7) * 8;       // n start
        const __bf16* src = Bw + (size_t)(k0 + r) * N + n0 + c;
        __builtin_prefetch(src + 32 * N, 0, 3);
        alignas(16) __bf16 tmp[8];
        *(uint4*)tmp = *(const uint4*)src;
#pragma unroll
        for (int i = 0; i < 8; i++) Bs[buf][(c + i) * 40 + r] = tmp[i];
    };

    // ---- prologue: stage buffer 0 ------------------------------------------
    const int NKT = K / 32;
    if (wave == 0) issue_A(0, 0);
    fill_B(0, 0);
    if (wave == 0) __builtin_amdgcn_s_wait_tensorcnt(0);
    __syncthreads();

    // ---- main pipeline -----------------------------------------------------
    for (int kt = 0; kt < NKT; kt++) {
        const int cur = kt & 1;
        if (kt + 1 < NKT) {                       // stage next tile (overlaps WMMA)
            if (wave == 0) issue_A(cur ^ 1, (kt + 1) * 32);
            fill_B(cur ^ 1, (kt + 1) * 32);
        }

        FragU afr[2], bfr[2];
#pragma unroll
        for (int mt = 0; mt < 2; mt++) {
            int mloc = wm * 32 + mt * 16 + l16;
#pragma unroll
            for (int i = 0; i < 8; i++) {
                int kb = (i < 4) ? (half * 8 + 2 * i) : (16 + half * 8 + 2 * (i - 4));
                afr[mt].u[i] = *(const unsigned int*)&As[cur][mloc * 40 + kb];
            }
        }
#pragma unroll
        for (int nt = 0; nt < 2; nt++) {
            int nloc = wn * 32 + nt * 16 + l16;
#pragma unroll
            for (int i = 0; i < 8; i++) {
                int kb = (i < 4) ? (half * 8 + 2 * i) : (16 + half * 8 + 2 * (i - 4));
                bfr[nt].u[i] = *(const unsigned int*)&Bs[cur][nloc * 40 + kb];
            }
        }
#pragma unroll
        for (int mt = 0; mt < 2; mt++)
#pragma unroll
            for (int nt = 0; nt < 2; nt++)
                acc[mt][nt] = __builtin_amdgcn_wmma_f32_16x16x32_bf16(
                    false, afr[mt].v, false, bfr[nt].v,
                    (short)0, acc[mt][nt], false, false);

        if (wave == 0) __builtin_amdgcn_s_wait_tensorcnt(0);  // next A tile landed
        __syncthreads();                                      // Bs stores + hand-off
    }

    // ---- epilogue: D layout VGPR r -> M = r + 8*half, N = lane&15
#pragma unroll
    for (int mt = 0; mt < 2; mt++)
#pragma unroll
        for (int nt = 0; nt < 2; nt++) {
            int gn = n0 + wn * 32 + nt * 16 + l16;
            float bval = bias[gn];
#pragma unroll
            for (int r = 0; r < 8; r++) {
                int gm = m0 + wm * 32 + mt * 16 + r + 8 * half;
                C[(size_t)gm * N + gn] = acc[mt][nt][r] + bval;
            }
        }
}

// ---------------------------------------------------------------------------
// Depthwise 5x5 conv on v (qkvo slice 2C..3C), output NHWC -> lepe[b,n,c]
// ---------------------------------------------------------------------------
__global__ void lepe_conv_kernel(const float* __restrict__ qkvo,
                                 const float* __restrict__ w,   // [C][5][5]
                                 const float* __restrict__ bconv,
                                 float* __restrict__ lepe)
{
    int idx = blockIdx.x * blockDim.x + threadIdx.x;   // over B*N*C
    if (idx >= BB * NN * CC) return;
    int c   = idx & (CC - 1);
    int row = idx >> 8;           // b*N + n
    int n   = row & (NN - 1);
    int b   = row >> 12;
    int h   = n >> 6, wpx = n & 63;
    float acc = bconv[c];
    const float* wc = w + c * 25;
#pragma unroll
    for (int kh = 0; kh < 5; kh++) {
        int ih = h + kh - 2;
        if (ih < 0 || ih >= HIMG) continue;
#pragma unroll
        for (int kw = 0; kw < 5; kw++) {
            int iw = wpx + kw - 2;
            if (iw < 0 || iw >= WIMG) continue;
            float vv = qkvo[((size_t)(b * NN + ih * 64 + iw)) * QKVO_LD + 2 * CC + c];
            acc += vv * wc[kh * 5 + kw];
        }
    }
    lepe[idx] = acc;
}

// ---------------------------------------------------------------------------
// RALA attention core: one workgroup per (b,h). 256 threads.
// ---------------------------------------------------------------------------
__global__ __launch_bounds__(256) void rala_attn_kernel(
    const float* __restrict__ qkvo, float* __restrict__ attn)
{
    const int bh = blockIdx.x;
    const int b  = bh >> 3;
    const int h  = bh & 7;
    const float* base = qkvo + (size_t)b * NN * QKVO_LD + h * HD;
    const int tid = threadIdx.x;

    __shared__ float s[NN];            // logits -> eff
    __shared__ float red[256];
    __shared__ float qmean[HD];
    __shared__ float kmean[HD];
    __shared__ float kmpart[16][HD];
    __shared__ float kr[64][34];       // rope'd scaled k chunk
    __shared__ float vv[64][34];       // v chunk
    __shared__ float kvs[HD][33];      // kv 32x32
    __shared__ float thetas[8];

    if (tid < 8) thetas[tid] = __powf(10000.f, -(float)tid * 0.125f);
    __syncthreads();

    // ---------------- Phase A: q_mean ----------------
    {
        int d = tid & 31, n0 = tid >> 5;
        float acc = 0.f;
        for (int n = n0; n < NN; n += 8)
            acc += elu1(base[(size_t)n * QKVO_LD + d]);
        red[tid] = acc;
        __syncthreads();
        if (tid < HD) {
            float sm = 0.f;
#pragma unroll
            for (int i = 0; i < 8; i++) sm += red[i * 32 + tid];
            qmean[tid] = sm * (1.f / NN);
        }
        __syncthreads();
    }

    // ---------------- Phase B/C: softmax eff over n ----------------
    const float scale = 0.1767766952966369f;    // HD^-0.5
    float lmax = -3.0e38f;
    for (int n = tid; n < NN; n += 256) {
        const float* kp = base + (size_t)n * QKVO_LD + CC;
        float dot = 0.f;
#pragma unroll
        for (int d = 0; d < HD; d++) dot += qmean[d] * elu1(kp[d]);
        float sv = dot * scale;
        s[n] = sv;
        lmax = fmaxf(lmax, sv);
    }
    red[tid] = lmax; __syncthreads();
    for (int off = 128; off > 0; off >>= 1) {
        if (tid < off) red[tid] = fmaxf(red[tid], red[tid + off]);
        __syncthreads();
    }
    float gmax = red[0]; __syncthreads();
    float lsum = 0.f;
    for (int n = tid; n < NN; n += 256) {
        float e = __expf(s[n] - gmax);
        s[n] = e;
        lsum += e;
    }
    red[tid] = lsum; __syncthreads();
    for (int off = 128; off > 0; off >>= 1) {
        if (tid < off) red[tid] += red[tid + off];
        __syncthreads();
    }
    float invsum = 1.f / red[0]; __syncthreads();
    for (int n = tid; n < NN; n += 256) s[n] *= invsum;     // eff_n
    __syncthreads();

    // ---------------- Phase D: kv = sum_n rope(elu(k)*eff) (x) v ; kmean ----
    const int p     = tid & 15;              // rope pair index per thread
    const float th  = thetas[p & 7];
    const bool useH = (p < 8);
    const int kd  = tid >> 3;                // kv row owned by thread
    const int ke0 = (tid & 7) * 4;           // kv cols [ke0, ke0+4)
    float kv0 = 0.f, kv1 = 0.f, kv2 = 0.f, kv3 = 0.f;
    float km0 = 0.f, km1 = 0.f;

    for (int chunk = 0; chunk < NN / 64; chunk++) {
        int rbase = chunk * 64;
#pragma unroll
        for (int j = 0; j < 4; j++) {
            int idx = tid + j * 256;
            int row = idx >> 4;              // 0..63
            int n = rbase + row;
            const float* kp = base + (size_t)n * QKVO_LD + CC;
            const float* vp = base + (size_t)n * QKVO_LD + 2 * CC;
            float eff = s[n];
            float k0 = elu1(kp[2 * p])     * eff;
            float k1 = elu1(kp[2 * p + 1]) * eff;
            km0 += k0; km1 += k1;
            float ang = (useH ? (float)(n >> 6) : (float)(n & 63)) * th;
            float sn, cs;
            __sincosf(ang, &sn, &cs);
            kr[row][2 * p]     = k0 * cs - k1 * sn;
            kr[row][2 * p + 1] = k0 * sn + k1 * cs;
            vv[row][2 * p]     = vp[2 * p];
            vv[row][2 * p + 1] = vp[2 * p + 1];
        }
        __syncthreads();
        for (int r = 0; r < 64; r++) {
            float krv = kr[r][kd];
            kv0 += krv * vv[r][ke0 + 0];
            kv1 += krv * vv[r][ke0 + 1];
            kv2 += krv * vv[r][ke0 + 2];
            kv3 += krv * vv[r][ke0 + 3];
        }
        __syncthreads();
    }
    kmpart[tid >> 4][2 * p]     = km0;
    kmpart[tid >> 4][2 * p + 1] = km1;
    kvs[kd][ke0 + 0] = kv0;
    kvs[kd][ke0 + 1] = kv1;
    kvs[kd][ke0 + 2] = kv2;
    kvs[kd][ke0 + 3] = kv3;
    __syncthreads();
    if (tid < HD) {
        float acc = 0.f;
#pragma unroll
        for (int i = 0; i < 16; i++) acc += kmpart[i][tid];
        kmean[tid] = acc;                    // sum_n elu(k)*eff  (== mean of scaled k)
    }
    __syncthreads();

    // ---------------- Phase E: out_n = rope(elu(q)) @ kv * z_n --------------
    for (int n = tid; n < NN; n += 256) {
        const float* qp = base + (size_t)n * QKVO_LD;
        float q[HD], qr[HD];
        float zdot = 0.f;
#pragma unroll
        for (int d = 0; d < HD; d++) {
            q[d] = elu1(qp[d]);
            zdot += q[d] * kmean[d];
        }
        float z = 1.f / (zdot + 1e-6f);
        float hh = (float)(n >> 6), ww = (float)(n & 63);
#pragma unroll
        for (int pp = 0; pp < 16; pp++) {
            float ang = (pp < 8 ? hh : ww) * thetas[pp & 7];
            float sn, cs;
            __sincosf(ang, &sn, &cs);
            qr[2 * pp]     = q[2 * pp] * cs - q[2 * pp + 1] * sn;
            qr[2 * pp + 1] = q[2 * pp] * sn + q[2 * pp + 1] * cs;
        }
        float* op = attn + ((size_t)(b * NN + n)) * CC + h * HD;
#pragma unroll
        for (int e = 0; e < HD; e++) {
            float acc = 0.f;
#pragma unroll
            for (int d = 0; d < HD; d++) acc += qr[d] * kvs[d][e];
            op[e] = acc * z;
        }
    }
}

// ---------------------------------------------------------------------------
// Fuse: t = (attn + lepe) * o  -> bf16
// ---------------------------------------------------------------------------
__global__ void fuse_kernel(const float* __restrict__ attn,
                            const float* __restrict__ lepe,
                            const float* __restrict__ qkvo,
                            __bf16* __restrict__ tb)
{
    int idx = blockIdx.x * blockDim.x + threadIdx.x;
    if (idx >= BB * NN * CC) return;
    int c   = idx & (CC - 1);
    int row = idx >> 8;
    float o = qkvo[(size_t)row * QKVO_LD + 3 * CC + c];
    tb[idx] = (__bf16)((attn[idx] + lepe[idx]) * o);
}

// ---------------------------------------------------------------------------
// Host launcher
// ---------------------------------------------------------------------------
extern "C" void kernel_launch(void* const* d_in, const int* in_sizes, int n_in,
                              void* d_out, int out_size, void* d_ws, size_t ws_size,
                              hipStream_t stream) {
    const float* x      = (const float*)d_in[0];   // [B,N,C]
    const float* w_qkvo = (const float*)d_in[1];   // [C,4C]
    const float* b_qkvo = (const float*)d_in[2];   // [4C]
    const float* lepe_w = (const float*)d_in[3];   // [C,1,5,5]
    const float* lepe_b = (const float*)d_in[4];   // [C]
    const float* w_proj = (const float*)d_in[5];   // [C,C]
    const float* b_proj = (const float*)d_in[6];   // [C]
    float* out = (float*)d_out;

    // workspace layout (256B-aligned offsets)
    char* ws = (char*)d_ws;
    size_t off = 0;
    auto carve = [&](size_t bytes) {
        size_t o = off;
        off += (bytes + 255) & ~(size_t)255;
        return o;
    };
    const size_t nXC   = (size_t)MROWS * CC;          // 16.7M
    const size_t nQKVO = (size_t)MROWS * QKVO_LD;     // 67.1M
    __bf16* x_bf   = (__bf16*)(ws + carve(nXC * 2));
    __bf16* wq_bf  = (__bf16*)(ws + carve((size_t)CC * 4 * CC * 2));
    __bf16* wp_bf  = (__bf16*)(ws + carve((size_t)CC * CC * 2));
    float*  qkvo   = (float*)(ws + carve(nQKVO * 4));
    float*  lepe   = (float*)(ws + carve(nXC * 4));
    float*  attn   = (float*)(ws + carve(nXC * 4));
    __bf16* t_bf   = (__bf16*)(ws + carve(nXC * 2));

    // 1) convert inputs to bf16
    cvt_bf16_kernel<<<(int)((nXC + 255) / 256), 256, 0, stream>>>(x, x_bf, (int)nXC);
    cvt_bf16_kernel<<<(CC * 4 * CC + 255) / 256, 256, 0, stream>>>(w_qkvo, wq_bf, CC * 4 * CC);
    cvt_bf16_kernel<<<(CC * CC + 255) / 256, 256, 0, stream>>>(w_proj, wp_bf, CC * CC);

    // 2) qkvo = x @ w_qkvo + b_qkvo     (M=65536, N=1024, K=256)
    gemm_bf16_wmma<<<dim3(QKVO_LD / 64, MROWS / 128), 256, 0, stream>>>(
        x_bf, wq_bf, b_qkvo, qkvo, MROWS, QKVO_LD, CC);

    // 3) LEPE depthwise conv (NHWC output)
    lepe_conv_kernel<<<(int)((nXC + 255) / 256), 256, 0, stream>>>(
        qkvo, lepe_w, lepe_b, lepe);

    // 4) attention core, one block per (b, head)
    rala_attn_kernel<<<BB * NHEAD, 256, 0, stream>>>(qkvo, attn);

    // 5) fuse (attn + lepe) * o -> bf16
    fuse_kernel<<<(int)((nXC + 255) / 256), 256, 0, stream>>>(attn, lepe, qkvo, t_bf);

    // 6) out = t @ w_proj + b_proj      (M=65536, N=256, K=256)
    gemm_bf16_wmma<<<dim3(CC / 64, MROWS / 128), 256, 0, stream>>>(
        t_bf, wp_bf, b_proj, out, MROWS, CC, CC);

    (void)in_sizes; (void)n_in; (void)out_size; (void)ws_size;
}
